// QuantizedLinear_67860483277527
// MI455X (gfx1250) — matrix operand order repaired and checked by reading, not compiled, not run on hardware
//
#include <hip/hip_runtime.h>
#include <stdint.h>

// ---------------------------------------------------------------------------
// QuantizedLinear on MI455X (gfx1250): out = (x @ wq^T) * scale + bias
//   x: [M,K] fp32, wq: [N,K] int8, scale/bias: [N] fp32, out: [M,N] fp32
//
// Compute-bound (AI ~1430 FLOP/B). Strategy:
//   pass 1: convert x -> f16 (32MB) and wq -> f16 (128MB) into d_ws
//           (memory-bound, ~7us; the 160MB working set fits the 192MB L2)
//   pass 2: f16 WMMA GEMM, 128x128 block tile, 8 waves of 32x64 each
//           (64 acc VGPRs/wave, fragments loaded en bloc so the allocator
//            keeps accumulators in place), triple-buffered LDS fed by
//           GLOBAL_LOAD_ASYNC_TO_LDS_B128 at prefetch distance 2.
// ---------------------------------------------------------------------------

typedef __attribute__((ext_vector_type(16))) _Float16 v16h;
typedef __attribute__((ext_vector_type(8)))  float    v8f;

constexpr int GM = 4096;
constexpr int GK = 4096;
constexpr int GN = 16384;

// --- async-to-LDS availability --------------------------------------------
#if defined(__has_builtin)
#  if __has_builtin(__builtin_amdgcn_global_load_async_to_lds_b128)
#    define HAS_ASYNC_LDS 1
#  endif
#endif
#ifndef HAS_ASYNC_LDS
#  define HAS_ASYNC_LDS 0
#endif

#if HAS_ASYNC_LDS
// Builtin signature (from clang diagnostic): param0 = as1 "int __vector(4)"*,
// param1 = as3 counterpart, then imm offset, imm cpol.
typedef int av4i __attribute__((vector_size(16)));
typedef __attribute__((address_space(1))) av4i gav4i;
typedef __attribute__((address_space(3))) av4i lav4i;
#define ASYNC_SRC(p) ((gav4i*)(uintptr_t)(p))
#define ASYNC_DST(p) ((lav4i*)(unsigned)(uintptr_t)(p))

__device__ __forceinline__ void async_wait_le4() {
#if __has_builtin(__builtin_amdgcn_s_wait_asynccnt)
  __builtin_amdgcn_s_wait_asynccnt(4);
#else
  asm volatile("s_wait_asynccnt 0x4" ::: "memory");
#endif
}
__device__ __forceinline__ void async_wait_le0() {
#if __has_builtin(__builtin_amdgcn_s_wait_asynccnt)
  __builtin_amdgcn_s_wait_asynccnt(0);
#else
  asm volatile("s_wait_asynccnt 0x0" ::: "memory");
#endif
}
#endif

union Frag { uint4 u[2]; v16h v; };

// ===========================================================================
// Pass-1 kernels: one-shot precision conversion into workspace
// ===========================================================================
__global__ __launch_bounds__(256) void cvt_x_to_f16(
    const float* __restrict__ in, _Float16* __restrict__ out) {
  const size_t i = (size_t)blockIdx.x * 256 + threadIdx.x;  // 8 floats/thread
  const float4* p = (const float4*)in + 2 * i;
  const float4 a = p[0], b = p[1];
  __align__(16) _Float16 h[8] = {
    (_Float16)a.x, (_Float16)a.y, (_Float16)a.z, (_Float16)a.w,
    (_Float16)b.x, (_Float16)b.y, (_Float16)b.z, (_Float16)b.w };
  ((uint4*)out)[i] = *(const uint4*)h;
}

__global__ __launch_bounds__(256) void cvt_w_to_f16(
    const signed char* __restrict__ in, _Float16* __restrict__ out) {
  const size_t i = (size_t)blockIdx.x * 256 + threadIdx.x;  // 16 bytes/thread
  const int4 v = ((const int4*)in)[i];
  const int vv[4] = { v.x, v.y, v.z, v.w };
  __align__(16) _Float16 h[16];
  #pragma unroll
  for (int j = 0; j < 4; ++j) {
    const int w = vv[j];
    h[j*4+0] = (_Float16)(float)(signed char)(w);
    h[j*4+1] = (_Float16)(float)(signed char)(w >> 8);
    h[j*4+2] = (_Float16)(float)(signed char)(w >> 16);
    h[j*4+3] = (_Float16)(float)(signed char)(w >> 24);
  }
  ((uint4*)out)[2*i]   = ((const uint4*)h)[0];
  ((uint4*)out)[2*i+1] = ((const uint4*)h)[1];
}

// ===========================================================================
// Pass-2 kernel: pure-f16 WMMA GEMM + dequant epilogue
// ===========================================================================
constexpr int BM  = 128;
constexpr int BN  = 128;
constexpr int BK  = 32;       // == WMMA K
constexpr int LDA = 40;       // padded LDS row stride (halves); 80B, 16B-aligned
constexpr int NK  = GK / BK;

__global__ __launch_bounds__(256) void gemm_f16_wmma(
    const _Float16* __restrict__ xh,   // [GM,GK] f16
    const _Float16* __restrict__ wh,   // [GN,GK] f16
    const float* __restrict__ wscale,
    const float* __restrict__ bias,
    float* __restrict__ out)
{
  __shared__ __align__(16) _Float16 As[3][BM * LDA];   // 30 KB
  __shared__ __align__(16) _Float16 Bs[3][BN * LDA];   // 30 KB

  const int tid  = threadIdx.x;
  const int lane = tid & 31;
  const int wave = tid >> 5;
  const int wm   = wave & 3;     // 4 waves along M -> 32 rows each
  const int wn   = wave >> 2;    // 2 waves along N -> 64 cols each

  const int blockM = blockIdx.y * BM;
  const int blockN = blockIdx.x * BN;

  // Copy-in mapping: 16B chunks; A tile = 512 chunks, B tile = 512 chunks.
  // chunk c: row = c>>2, q = c&3  (q indexes 16B within the 64B K-row)
#if HAS_ASYNC_LDS
  auto stage = [&](int kt, int buf) {
    const int k0 = kt * BK;
    #pragma unroll
    for (int i = 0; i < 2; ++i) {                 // A: 2 chunks/thread
      const int c = tid + 256 * i, row = c >> 2, q = c & 3;
      const _Float16* g = xh + (size_t)(blockM + row) * GK + k0 + q * 8;
      __builtin_amdgcn_global_load_async_to_lds_b128(
          ASYNC_SRC(g), ASYNC_DST(&As[buf][row * LDA + q * 8]), 0, 0);
    }
    #pragma unroll
    for (int i = 0; i < 2; ++i) {                 // B: 2 chunks/thread
      const int c = tid + 256 * i, row = c >> 2, q = c & 3;
      const _Float16* g = wh + (size_t)(blockN + row) * GK + k0 + q * 8;
      __builtin_amdgcn_global_load_async_to_lds_b128(
          ASYNC_SRC(g), ASYNC_DST(&Bs[buf][row * LDA + q * 8]), 0, 0);
    }
  };
#else
  uint4 ar[2], br[2];
  auto load_regs = [&](int kt) {
    const int k0 = kt * BK;
    #pragma unroll
    for (int i = 0; i < 2; ++i) {
      const int c = tid + 256 * i, row = c >> 2, q = c & 3;
      ar[i] = *(const uint4*)(xh + (size_t)(blockM + row) * GK + k0 + q * 8);
      br[i] = *(const uint4*)(wh + (size_t)(blockN + row) * GK + k0 + q * 8);
    }
  };
  auto store_regs = [&](int buf) {
    #pragma unroll
    for (int i = 0; i < 2; ++i) {
      const int c = tid + 256 * i, row = c >> 2, q = c & 3;
      *(uint4*)&As[buf][row * LDA + q * 8] = ar[i];
      *(uint4*)&Bs[buf][row * LDA + q * 8] = br[i];
    }
  };
#endif

  // Accumulators: 2(M) x 4(N) tiles of 16x16 f32 (64 VGPRs)
  v8f acc[2][4];
  #pragma unroll
  for (int tm = 0; tm < 2; ++tm)
    #pragma unroll
    for (int tn = 0; tn < 4; ++tn)
      acc[tm][tn] = {};

  // Fragment lane mapping (ISA 7.12.2, 16-bit operands, wave32)
  const int r     = lane & 15;
  const int grp   = lane >> 4;
  const int kselA = grp * 8;     // A: K in {g*8..g*8+7} and {16+g*8..}
  const int kselB = grp * 8;     // B: K in {g*8..g*8+7} and {16+g*8..}

  // Fragments loaded en bloc (round-1 structure): keeps accumulators pinned,
  // no acc<->fragment register recycling, no v_mov/v_nop tax around WMMAs.
  auto compute = [&](int cur) {
    Frag af[2], bf[4];
    #pragma unroll
    for (int tm = 0; tm < 2; ++tm) {
      const _Float16* p = &As[cur][(wm * 32 + tm * 16 + r) * LDA + kselA];
      af[tm].u[0] = *(const uint4*)p;
      af[tm].u[1] = *(const uint4*)(p + 16);
    }
    #pragma unroll
    for (int tn = 0; tn < 4; ++tn) {
      const _Float16* p = &Bs[cur][(wn * 64 + tn * 16 + r) * LDA + kselB];
      bf[tn].u[0] = *(const uint4*)p;
      bf[tn].u[1] = *(const uint4*)(p + 16);
    }
    #pragma unroll
    for (int tm = 0; tm < 2; ++tm)
      #pragma unroll
      for (int tn = 0; tn < 4; ++tn)
        acc[tm][tn] = __builtin_amdgcn_wmma_f32_16x16x32_f16(
            false, af[tm].v, false, bf[tn].v,
            (short)0, acc[tm][tn], false, false);
  };

#if HAS_ASYNC_LDS
  // Prefetch distance 2, 3 LDS buffers. Async loads complete in order, so
  // waiting ASYNCcnt<=4 (the 4 ops of the newest tile) proves the older
  // tile has fully landed; compute never exposes DMA latency.
  stage(0, 0);
  stage(1, 1);
  async_wait_le4();
  __syncthreads();
  int cur = 0;
  for (int kt = 0; kt < NK; ++kt) {
    if (kt + 2 < NK) {
      int b2 = cur + 2; if (b2 >= 3) b2 -= 3;
      stage(kt + 2, b2);
    }
    compute(cur);
    if (kt + 2 < NK) async_wait_le4();
    else             async_wait_le0();
    __syncthreads();
    ++cur; if (cur == 3) cur = 0;
  }
#else
  load_regs(0);
  store_regs(0);
  __syncthreads();
  for (int kt = 0; kt < NK; ++kt) {
    const int cur = kt & 1;
    if (kt + 1 < NK) load_regs(kt + 1);
    compute(cur);
    if (kt + 1 < NK) store_regs(cur ^ 1);
    __syncthreads();
  }
#endif

  // Epilogue: per-channel dequant + bias.
  // C layout: lanes 0-15 -> N=lane, M=vgpr i; lanes 16-31 -> N=lane-16, M=8+i
  const int nlo = lane & 15;
  const int mhi = (lane >> 4) * 8;
  #pragma unroll
  for (int tn = 0; tn < 4; ++tn) {
    const int gn = blockN + wn * 64 + tn * 16 + nlo;
    const float s = wscale[gn];
    const float b = bias[gn];
    #pragma unroll
    for (int tm = 0; tm < 2; ++tm) {
      const int gm0 = blockM + wm * 32 + tm * 16 + mhi;
      #pragma unroll
      for (int i = 0; i < 8; ++i)
        out[(size_t)(gm0 + i) * GN + gn] = acc[tm][tn][i] * s + b;
    }
  }
}

// ===========================================================================
// Fallback: fused kernel (conversion in-loop) if d_ws is too small
// ===========================================================================
constexpr int FBM = 128, FBN = 128, FLD = 40;

__global__ __launch_bounds__(256) void qlinear_fused(
    const float* __restrict__ x, const signed char* __restrict__ wq,
    const float* __restrict__ wscale, const float* __restrict__ bias,
    float* __restrict__ out)
{
  __shared__ __align__(16) _Float16 As[2][FBM * FLD];
  __shared__ __align__(16) _Float16 Bs[2][FBN * FLD];

  const int tid = threadIdx.x, lane = tid & 31, wave = tid >> 5;
  const int wm = wave & 3, wn = wave >> 2;
  const int blockM = blockIdx.y * FBM, blockN = blockIdx.x * FBN;
  const int xrow = tid >> 3, xcol4 = tid & 7;
  const int wrow = tid >> 1, whalf = tid & 1;
  const float* xg = x + (size_t)(blockM + xrow) * GK + xcol4 * 4;
  const signed char* wg = wq + (size_t)(blockN + wrow) * GK + whalf * 16;
  float4 xr[4]; int4 wr;

  auto load_global = [&](int kt) {
    const int k0 = kt * BK;
    #pragma unroll
    for (int i = 0; i < 4; ++i)
      xr[i] = *(const float4*)(xg + (size_t)(i * 32) * GK + k0);
    wr = *(const int4*)(wg + k0);
  };
  auto store_lds = [&](int buf) {
    #pragma unroll
    for (int i = 0; i < 4; ++i) {
      __align__(8) _Float16 h4[4] = {
        (_Float16)xr[i].x, (_Float16)xr[i].y,
        (_Float16)xr[i].z, (_Float16)xr[i].w };
      *(uint2*)&As[buf][(xrow + i * 32) * FLD + xcol4 * 4] = *(const uint2*)h4;
    }
    __align__(16) _Float16 hb[16];
    const int vv[4] = { wr.x, wr.y, wr.z, wr.w };
    #pragma unroll
    for (int j = 0; j < 4; ++j) {
      const int v = vv[j];
      hb[j*4+0] = (_Float16)(float)(signed char)(v);
      hb[j*4+1] = (_Float16)(float)(signed char)(v >> 8);
      hb[j*4+2] = (_Float16)(float)(signed char)(v >> 16);
      hb[j*4+3] = (_Float16)(float)(signed char)(v >> 24);
    }
    uint4* dst = (uint4*)&Bs[buf][wrow * FLD + whalf * 16];
    dst[0] = ((const uint4*)hb)[0];
    dst[1] = ((const uint4*)hb)[1];
  };

  v8f acc[2][4];
  #pragma unroll
  for (int tm = 0; tm < 2; ++tm)
    #pragma unroll
    for (int tn = 0; tn < 4; ++tn) acc[tm][tn] = {};

  const int r = lane & 15, grp = lane >> 4;
  const int kselA = grp * 8, kselB = grp * 8;

  load_global(0); store_lds(0); __syncthreads();
  for (int kt = 0; kt < NK; ++kt) {
    const int cur = kt & 1;
    if (kt + 1 < NK) load_global(kt + 1);
    Frag af[2], bf[4];
    #pragma unroll
    for (int tm = 0; tm < 2; ++tm) {
      const _Float16* p = &As[cur][(wm * 32 + tm * 16 + r) * FLD + kselA];
      af[tm].u[0] = *(const uint4*)p;
      af[tm].u[1] = *(const uint4*)(p + 16);
    }
    #pragma unroll
    for (int tn = 0; tn < 4; ++tn) {
      const _Float16* p = &Bs[cur][(wn * 64 + tn * 16 + r) * FLD + kselB];
      bf[tn].u[0] = *(const uint4*)p;
      bf[tn].u[1] = *(const uint4*)(p + 16);
    }
    #pragma unroll
    for (int tm = 0; tm < 2; ++tm)
      #pragma unroll
      for (int tn = 0; tn < 4; ++tn)
        acc[tm][tn] = __builtin_amdgcn_wmma_f32_16x16x32_f16(
            false, af[tm].v, false, bf[tn].v, (short)0, acc[tm][tn], false, false);
    if (kt + 1 < NK) store_lds(cur ^ 1);
    __syncthreads();
  }

  const int nlo = lane & 15, mhi = (lane >> 4) * 8;
  #pragma unroll
  for (int tn = 0; tn < 4; ++tn) {
    const int gn = blockN + wn * 64 + tn * 16 + nlo;
    const float s = wscale[gn], b = bias[gn];
    #pragma unroll
    for (int tm = 0; tm < 2; ++tm) {
      const int gm0 = blockM + wm * 32 + tm * 16 + mhi;
      #pragma unroll
      for (int i = 0; i < 8; ++i)
        out[(size_t)(gm0 + i) * GN + gn] = acc[tm][tn][i] * s + b;
    }
  }
}

// ===========================================================================
extern "C" void kernel_launch(void* const* d_in, const int* in_sizes, int n_in,
                              void* d_out, int out_size, void* d_ws, size_t ws_size,
                              hipStream_t stream) {
  const float*       x  = (const float*)d_in[0];
  const signed char* wq = (const signed char*)d_in[1];
  const float*       ws = (const float*)d_in[2];
  const float*       b  = (const float*)d_in[3];
  float*             o  = (float*)d_out;

  const size_t need = ((size_t)GM * GK + (size_t)GN * GK) * sizeof(_Float16); // 160 MB
  if (ws_size >= need) {
    _Float16* xh = (_Float16*)d_ws;
    _Float16* wh = xh + (size_t)GM * GK;
    cvt_x_to_f16<<<(GM * (size_t)GK) / (8 * 256), 256, 0, stream>>>(x, xh);
    cvt_w_to_f16<<<((size_t)GN * GK) / (16 * 256), 256, 0, stream>>>(wq, wh);
    dim3 grid(GN / BN, GM / BM);   // (128, 32)
    gemm_f16_wmma<<<grid, 256, 0, stream>>>(xh, wh, ws, b, o);
  } else {
    dim3 grid(GN / FBN, GM / FBM); // (128, 32)
    qlinear_fused<<<grid, 256, 0, stream>>>(x, wq, ws, b, o);
  }
}
